// TransformerEncoderLayer_42855183679892
// MI455X (gfx1250) — compile-verified, matrix operand order checked by reading, not run on hardware
//
#include <hip/hip_runtime.h>

#define NHEAD 16
#define SEQ_T 2048
#define BATCH 2
#define NROWS (BATCH * SEQ_T)       /* 4096 token rows */
#define DM    1024
#define FFH   4096
#define ALPHA_C 1.4142135f

typedef __attribute__((ext_vector_type(16))) __bf16        bf16x16;
typedef __attribute__((ext_vector_type(8)))  float         f32x8;
typedef __attribute__((ext_vector_type(4)))  unsigned int  u32x4;

__device__ inline unsigned short f2bf(float f) {
  union { float f; unsigned u; } c; c.f = f;
  unsigned r = c.u + 0x7FFFu + ((c.u >> 16) & 1u);
  return (unsigned short)(r >> 16);
}
__device__ inline float bf2f(unsigned short h) {
  union { unsigned u; float f; } c; c.u = ((unsigned)h) << 16;
  return c.f;
}

// Assemble a 16x32 bf16 A/B fragment (wave32 layout, ISA 7.12.2):
// per lane: K elements {lh*8..lh*8+7} and {lh*8+16..lh*8+23} -> two 16B chunks.
__device__ inline bf16x16 frag_ld(const unsigned short* p0, const unsigned short* p1) {
  union { u32x4 u[2]; bf16x16 v; } t;
  t.u[0] = *(const u32x4*)p0;
  t.u[1] = *(const u32x4*)p1;
  return t.v;
}

__device__ inline f32x8 wmma_bf16(bf16x16 a, bf16x16 b, f32x8 c) {
  return __builtin_amdgcn_wmma_f32_16x16x32_bf16(false, a, false, b, (short)0, c, false, false);
}

// ---------------------------------------------------------------- cast
__global__ __launch_bounds__(256) void cast_f32_to_bf16(
    const float* __restrict__ in, unsigned short* __restrict__ out, int n) {
  int i = blockIdx.x * 256 + threadIdx.x;
  if (i < n) out[i] = f2bf(in[i]);
}

// ---------------------------------------------------------------- GEMM  C = A * B^T
// A[M,K] bf16, B[N,K] bf16.  MODE 0: f32 out, 1: f32 out + bias, 2: bf16 out.
// Block tile 128x128, BK=32, 8 waves, wave tile 64x32 (4x2 WMMA tiles).
// Double-buffered LDS: global loads for step k+1 overlap WMMA of step k.
template <int MODE>
__global__ __launch_bounds__(256) void gemm_nt(
    const unsigned short* __restrict__ A, const unsigned short* __restrict__ Bm,
    void* __restrict__ C, const float* __restrict__ bias, int M, int N, int K) {
  __shared__ unsigned short Ash[2][128 * 48];
  __shared__ unsigned short Bsh[2][128 * 48];
  const int tid = threadIdx.x;
  const int m0 = blockIdx.y * 128;
  const int n0 = blockIdx.x * 128;
  const int wid = tid >> 5, lane = tid & 31;
  const int wm = (wid >> 2) * 64, wn = (wid & 3) * 32;
  const int lh = lane >> 4, ln = lane & 15;
  const int kb = lh * 8;
  const int lrow = tid >> 1, lcol = (tid & 1) * 16;   // 2 chunks of 8 bf16 / thread / matrix
  f32x8 acc[4][2] = {};

  u32x4 ra0, ra1, rb0, rb1;
  auto gload = [&](int k0) {
    const unsigned short* ap = &A[(size_t)(m0 + lrow) * K + k0 + lcol];
    const unsigned short* bp = &Bm[(size_t)(n0 + lrow) * K + k0 + lcol];
    ra0 = *(const u32x4*)ap;
    ra1 = *(const u32x4*)(ap + 8);
    rb0 = *(const u32x4*)bp;
    rb1 = *(const u32x4*)(bp + 8);
  };
  auto lstore = [&](int buf) {
    *(u32x4*)&Ash[buf][lrow * 48 + lcol]     = ra0;
    *(u32x4*)&Ash[buf][lrow * 48 + lcol + 8] = ra1;
    *(u32x4*)&Bsh[buf][lrow * 48 + lcol]     = rb0;
    *(u32x4*)&Bsh[buf][lrow * 48 + lcol + 8] = rb1;
  };

  const int nk = K >> 5;
  gload(0);
  lstore(0);
  __syncthreads();
  for (int kt = 0; kt < nk; kt++) {
    const int cur = kt & 1;
    if (kt + 1 < nk) gload((kt + 1) << 5);
    bf16x16 afr[4], bfr[2];
#pragma unroll
    for (int i = 0; i < 4; i++)
      afr[i] = frag_ld(&Ash[cur][(wm + i * 16 + ln) * 48 + kb],
                       &Ash[cur][(wm + i * 16 + ln) * 48 + kb + 16]);
#pragma unroll
    for (int j = 0; j < 2; j++)
      bfr[j] = frag_ld(&Bsh[cur][(wn + j * 16 + ln) * 48 + kb],
                       &Bsh[cur][(wn + j * 16 + ln) * 48 + kb + 16]);
#pragma unroll
    for (int i = 0; i < 4; i++)
#pragma unroll
      for (int j = 0; j < 2; j++)
        acc[i][j] = wmma_bf16(afr[i], bfr[j], acc[i][j]);
    if (kt + 1 < nk) lstore(cur ^ 1);
    __syncthreads();
  }
#pragma unroll
  for (int i = 0; i < 4; i++)
#pragma unroll
    for (int j = 0; j < 2; j++) {
      int col = n0 + wn + j * 16 + ln;
      float bv = (MODE == 1) ? bias[col] : 0.f;
#pragma unroll
      for (int r = 0; r < 8; r++) {
        int row = m0 + wm + i * 16 + 8 * lh + r;  // C-layout: M = r + 8*(lane>=16)
        float v = acc[i][j][r] + bv;
        size_t o = (size_t)row * N + col;
        if (MODE == 2) ((unsigned short*)C)[o] = f2bf(v);
        else           ((float*)C)[o] = v;
      }
    }
}

// ---------------------------------------------------------------- dual-B GEMM + SwiGLU
// C[m,n] = y*g*sigmoid(g), y = A*B1^T, g = A*B2^T   (bf16 out)
// Block tile 128x128, wave tile 64x32 against both B matrices (16 WMMA / step).
__global__ __launch_bounds__(256) void gemm_swiglu(
    const unsigned short* __restrict__ A, const unsigned short* __restrict__ B1,
    const unsigned short* __restrict__ B2, unsigned short* __restrict__ C,
    int M, int N, int K) {
  __shared__ unsigned short Ash[128 * 48];
  __shared__ unsigned short Bs1[128 * 48];
  __shared__ unsigned short Bs2[128 * 48];
  const int tid = threadIdx.x;
  const int m0 = blockIdx.y * 128;
  const int n0 = blockIdx.x * 128;
  const int wid = tid >> 5, lane = tid & 31;
  const int wm = (wid >> 2) * 64, wn = (wid & 3) * 32;
  const int lh = lane >> 4, ln = lane & 15;
  const int kb = lh * 8;
  const int lrow = tid >> 1, lcol = (tid & 1) * 16;
  f32x8 accY[4][2] = {};
  f32x8 accG[4][2] = {};

  for (int k0 = 0; k0 < K; k0 += 32) {
    const unsigned short* ap = &A[(size_t)(m0 + lrow) * K + k0 + lcol];
    const unsigned short* p1 = &B1[(size_t)(n0 + lrow) * K + k0 + lcol];
    const unsigned short* p2 = &B2[(size_t)(n0 + lrow) * K + k0 + lcol];
    *(u32x4*)&Ash[lrow * 48 + lcol]     = *(const u32x4*)ap;
    *(u32x4*)&Ash[lrow * 48 + lcol + 8] = *(const u32x4*)(ap + 8);
    *(u32x4*)&Bs1[lrow * 48 + lcol]     = *(const u32x4*)p1;
    *(u32x4*)&Bs1[lrow * 48 + lcol + 8] = *(const u32x4*)(p1 + 8);
    *(u32x4*)&Bs2[lrow * 48 + lcol]     = *(const u32x4*)p2;
    *(u32x4*)&Bs2[lrow * 48 + lcol + 8] = *(const u32x4*)(p2 + 8);
    __syncthreads();
    bf16x16 afr[4], b1f[2], b2f[2];
#pragma unroll
    for (int i = 0; i < 4; i++)
      afr[i] = frag_ld(&Ash[(wm + i * 16 + ln) * 48 + kb],
                       &Ash[(wm + i * 16 + ln) * 48 + kb + 16]);
#pragma unroll
    for (int j = 0; j < 2; j++) {
      b1f[j] = frag_ld(&Bs1[(wn + j * 16 + ln) * 48 + kb],
                       &Bs1[(wn + j * 16 + ln) * 48 + kb + 16]);
      b2f[j] = frag_ld(&Bs2[(wn + j * 16 + ln) * 48 + kb],
                       &Bs2[(wn + j * 16 + ln) * 48 + kb + 16]);
    }
#pragma unroll
    for (int i = 0; i < 4; i++)
#pragma unroll
      for (int j = 0; j < 2; j++) {
        accY[i][j] = wmma_bf16(afr[i], b1f[j], accY[i][j]);
        accG[i][j] = wmma_bf16(afr[i], b2f[j], accG[i][j]);
      }
    __syncthreads();
  }
#pragma unroll
  for (int i = 0; i < 4; i++)
#pragma unroll
    for (int j = 0; j < 2; j++) {
      int col = n0 + wn + j * 16 + ln;
#pragma unroll
      for (int r = 0; r < 8; r++) {
        int row = m0 + wm + i * 16 + 8 * lh + r;
        float yv = accY[i][j][r];
        float gv = accG[i][j][r];
        float sg = 1.f / (1.f + __expf(-gv));
        C[(size_t)row * N + col] = f2bf(yv * gv * sg);
      }
    }
}

// ---------------------------------------------------------------- RoPE + qkv split
// qkv[NROWS,3072] bf16 -> q,k [B,H,T,64] (rotated), v transposed [B,H,64,T]
__global__ __launch_bounds__(256) void rope_split_kernel(
    const unsigned short* __restrict__ qkv,
    unsigned short* __restrict__ qb, unsigned short* __restrict__ kb,
    unsigned short* __restrict__ vt) {
  int idx = blockIdx.x * 256 + threadIdx.x;    // 2*2048*16*32 threads
  int i = idx & 31;
  int h = (idx >> 5) & 15;
  int t = (idx >> 9) & 2047;
  int b = idx >> 20;
  size_t base = ((size_t)(b * SEQ_T + t)) * 3072 + h * 64;
  float q1 = bf2f(qkv[base + i]),        q2 = bf2f(qkv[base + 32 + i]);
  float k1 = bf2f(qkv[base + 1024 + i]), k2 = bf2f(qkv[base + 1024 + 32 + i]);
  unsigned short v1 = qkv[base + 2048 + i], v2 = qkv[base + 2048 + 32 + i];
  float inv = __powf(10000.f, -(float)i * (1.f / 32.f));
  float sv, cv;
  __sincosf((float)t * inv, &sv, &cv);
  size_t qo = ((size_t)((b * NHEAD + h) * SEQ_T + t)) * 64;
  qb[qo + i]      = f2bf(q1 * cv - q2 * sv);
  qb[qo + 32 + i] = f2bf(q2 * cv + q1 * sv);
  kb[qo + i]      = f2bf(k1 * cv - k2 * sv);
  kb[qo + 32 + i] = f2bf(k2 * cv + k1 * sv);
  size_t vo = ((size_t)((b * NHEAD + h) * 64 + i)) * SEQ_T + t;
  vt[vo] = v1;
  vt[vo + (size_t)32 * SEQ_T] = v2;
}

// ---------------------------------------------------------------- banded flash attention
// one wave per 16-query tile; window rel in [-127,128]
__global__ __launch_bounds__(32) void attn_kernel(
    const unsigned short* __restrict__ qb, const unsigned short* __restrict__ kbuf,
    const unsigned short* __restrict__ vt, unsigned short* __restrict__ ob) {
  const int T = SEQ_T;
  int bid = blockIdx.x;
  int qt = bid & 127;
  int h = (bid >> 7) & 15;
  int b = bid >> 11;
  int q0 = qt * 16;
  int lane = threadIdx.x, lh = lane >> 4, ln = lane & 15;
  const int kb8 = lh * 8;
  const unsigned short* qhead = qb   + ((size_t)(b * NHEAD + h) * T) * 64;
  const unsigned short* khead = kbuf + ((size_t)(b * NHEAD + h) * T) * 64;
  const unsigned short* vhead = vt   + ((size_t)(b * NHEAD + h) * 64) * T;

  const unsigned short* qrow = qhead + (size_t)(q0 + ln) * 64;
  bf16x16 qf0 = frag_ld(qrow + kb8,      qrow + kb8 + 16);
  bf16x16 qf1 = frag_ld(qrow + 32 + kb8, qrow + 32 + kb8 + 16);

  f32x8 oacc[4] = {};
  float mrow[8], lrow[8];
#pragma unroll
  for (int r = 0; r < 8; r++) { mrow[r] = -1e30f; lrow[r] = 0.f; }
  __shared__ unsigned short Psh[16 * 48];
  const float scale = 0.125f;  // 1/sqrt(64)

  for (int p = 0; p < 9; p++) {
    int k0 = q0 - 128 + p * 32;           // 32-key slab
    // prefetch next slab's K rows into cache (global_prefetch_b8)
    {
      int pk = min(max(k0 + 32 + lane, 0), T - 1);
      __builtin_prefetch(khead + (size_t)pk * 64, 0, 3);
    }
    f32x8 s[2] = {};
#pragma unroll
    for (int t = 0; t < 2; t++) {
      int key = k0 + t * 16 + ln;
      int keyc = min(max(key, 0), T - 1); // clamp for safe loads; masked below
      const unsigned short* krow = khead + (size_t)keyc * 64;
      bf16x16 kf0 = frag_ld(krow + kb8,      krow + kb8 + 16);
      bf16x16 kf1 = frag_ld(krow + 32 + kb8, krow + 32 + kb8 + 16);
      s[t] = wmma_bf16(qf0, kf0, s[t]);
      s[t] = wmma_bf16(qf1, kf1, s[t]);
    }
    float p0[8], p1[8], ef[8];
#pragma unroll
    for (int r = 0; r < 8; r++) {
      int q = q0 + 8 * lh + r;
      int key0 = k0 + ln, key1 = k0 + 16 + ln;
      int r0 = key0 - q, r1 = key1 - q;
      bool v0 = (r0 >= -127) && (r0 <= 128) && (key0 >= 0) && (key0 < T);
      bool v1 = (r1 >= -127) && (r1 <= 128) && (key1 >= 0) && (key1 < T);
      float s0 = v0 ? s[0][r] * scale : -1e30f;
      float s1 = v1 ? s[1][r] * scale : -1e30f;
      float mx = fmaxf(s0, s1);
#pragma unroll
      for (int d = 1; d < 16; d <<= 1) mx = fmaxf(mx, __shfl_xor(mx, d, 32));
      float mnew = fmaxf(mrow[r], mx);
      ef[r] = __expf(mrow[r] - mnew);
      p0[r] = v0 ? __expf(s0 - mnew) : 0.f;
      p1[r] = v1 ? __expf(s1 - mnew) : 0.f;
      float ps = p0[r] + p1[r];
#pragma unroll
      for (int d = 1; d < 16; d <<= 1) ps += __shfl_xor(ps, d, 32);
      lrow[r] = lrow[r] * ef[r] + ps;
      mrow[r] = mnew;
    }
#pragma unroll
    for (int nd = 0; nd < 4; nd++)
#pragma unroll
      for (int r = 0; r < 8; r++) oacc[nd][r] *= ef[r];

    // P (16q x 32key) -> LDS, re-read as A-fragment
#pragma unroll
    for (int r = 0; r < 8; r++) {
      Psh[(8 * lh + r) * 48 + ln]      = f2bf(p0[r]);
      Psh[(8 * lh + r) * 48 + 16 + ln] = f2bf(p1[r]);
    }
    __syncthreads();
    bf16x16 pf = frag_ld(&Psh[ln * 48 + kb8], &Psh[ln * 48 + kb8 + 16]);
#pragma unroll
    for (int nd = 0; nd < 4; nd++) {
      int d = nd * 16 + ln;
      const unsigned short* vrow = vhead + (size_t)d * T;
      int c1 = min(max(k0 + kb8, 0), T - 8);
      int c2 = min(max(k0 + kb8 + 16, 0), T - 8);
      bf16x16 vf = frag_ld(vrow + c1, vrow + c2);
      oacc[nd] = wmma_bf16(pf, vf, oacc[nd]);
    }
    __syncthreads();
  }
#pragma unroll
  for (int r = 0; r < 8; r++) lrow[r] = 1.f / lrow[r];
#pragma unroll
  for (int nd = 0; nd < 4; nd++)
#pragma unroll
    for (int r = 0; r < 8; r++) {
      int row = q0 + 8 * lh + r;
      ob[((size_t)b * T + row) * DM + h * 64 + nd * 16 + ln] = f2bf(oacc[nd][r] * lrow[r]);
    }
}

// ---------------------------------------------------------------- residual + RMSNorm
template <bool WB>
__global__ __launch_bounds__(256) void resid_rms(
    const float* __restrict__ a, const float* __restrict__ x,
    const float* __restrict__ g, float* __restrict__ outf,
    unsigned short* __restrict__ outb) {
  __shared__ float red[8];
  int row = blockIdx.x, tid = threadIdx.x;
  const float* ar = a + (size_t)row * DM;
  const float* xr = x + (size_t)row * DM;
  float v[4];
  float ss = 0.f;
#pragma unroll
  for (int i = 0; i < 4; i++) {
    int c = tid + i * 256;
    v[i] = ar[c] + ALPHA_C * xr[c];
    ss += v[i] * v[i];
  }
#pragma unroll
  for (int d = 1; d < 32; d <<= 1) ss += __shfl_xor(ss, d, 32);
  if ((tid & 31) == 0) red[tid >> 5] = ss;
  __syncthreads();
  float tot = 0.f;
#pragma unroll
  for (int w = 0; w < 8; w++) tot += red[w];
  float rms = rsqrtf(tot * (1.f / (float)DM) + 1e-5f);
#pragma unroll
  for (int i = 0; i < 4; i++) {
    int c = tid + i * 256;
    float o = v[i] * rms * g[c];
    outf[(size_t)row * DM + c] = o;
    if (WB) outb[(size_t)row * DM + c] = f2bf(o);
  }
}

// ---------------------------------------------------------------- launch
extern "C" void kernel_launch(void* const* d_in, const int* in_sizes, int n_in,
                              void* d_out, int out_size, void* d_ws, size_t ws_size,
                              hipStream_t stream) {
  const float* x    = (const float*)d_in[0];
  const float* Wqkv = (const float*)d_in[1];
  const float* Wout = (const float*)d_in[2];
  const float* bout = (const float*)d_in[3];
  const float* W1   = (const float*)d_in[4];
  const float* W2   = (const float*)d_in[5];
  const float* g1   = (const float*)d_in[6];
  const float* g2   = (const float*)d_in[7];

  char* ws = (char*)d_ws;
  size_t off = 0;
  auto alloc = [&](size_t bytes) -> char* {
    char* p = ws + off;
    off += (bytes + 255) & ~(size_t)255;
    return p;
  };
  unsigned short* xb    = (unsigned short*)alloc((size_t)NROWS * DM * 2);
  unsigned short* wqkvb = (unsigned short*)alloc((size_t)3 * DM * DM * 2);
  unsigned short* woutb = (unsigned short*)alloc((size_t)DM * DM * 2);
  unsigned short* w1b   = (unsigned short*)alloc((size_t)2 * FFH * DM * 2);
  unsigned short* w2b   = (unsigned short*)alloc((size_t)DM * FFH * 2);
  unsigned short* qkvb  = (unsigned short*)alloc((size_t)NROWS * 3 * DM * 2);
  unsigned short* qbuf  = (unsigned short*)alloc((size_t)NROWS * DM * 2);
  unsigned short* kbuf  = (unsigned short*)alloc((size_t)NROWS * DM * 2);
  unsigned short* vtb   = (unsigned short*)alloc((size_t)NROWS * DM * 2);
  unsigned short* obuf  = (unsigned short*)alloc((size_t)NROWS * DM * 2);
  float*          af    = (float*)alloc((size_t)NROWS * DM * 4);
  float*          x2f   = (float*)alloc((size_t)NROWS * DM * 4);
  unsigned short* x2b   = (unsigned short*)alloc((size_t)NROWS * DM * 2);
  unsigned short* yb    = (unsigned short*)alloc((size_t)NROWS * FFH * 2);
  float*          y2f   = (float*)alloc((size_t)NROWS * DM * 4);

  auto castN = [&](const float* src, unsigned short* dst, int n) {
    cast_f32_to_bf16<<<(n + 255) / 256, 256, 0, stream>>>(src, dst, n);
  };
  castN(x,    xb,    NROWS * DM);
  castN(Wqkv, wqkvb, 3 * DM * DM);
  castN(Wout, woutb, DM * DM);
  castN(W1,   w1b,   2 * FFH * DM);
  castN(W2,   w2b,   DM * FFH);

  // qkv = x @ Wqkv^T  (bf16 out)
  gemm_nt<2><<<dim3(3 * DM / 128, NROWS / 128), 256, 0, stream>>>(
      xb, wqkvb, qkvb, nullptr, NROWS, 3 * DM, DM);

  rope_split_kernel<<<(BATCH * SEQ_T * NHEAD * 32) / 256, 256, 0, stream>>>(
      qkvb, qbuf, kbuf, vtb);

  attn_kernel<<<BATCH * NHEAD * (SEQ_T / 16), 32, 0, stream>>>(qbuf, kbuf, vtb, obuf);

  // a = o @ Wout^T + bout  (f32 out)
  gemm_nt<1><<<dim3(DM / 128, NROWS / 128), 256, 0, stream>>>(
      obuf, woutb, af, bout, NROWS, DM, DM);

  // x2 = rmsnorm(a + alpha*x, g1) -> f32 + bf16
  resid_rms<true><<<NROWS, 256, 0, stream>>>(af, x, g1, x2f, x2b);

  // y = swiglu(x2 @ W1a^T, x2 @ W1b^T)  (bf16 out)
  gemm_swiglu<<<dim3(FFH / 128, NROWS / 128), 256, 0, stream>>>(
      x2b, w1b, w1b + (size_t)FFH * DM, yb, NROWS, FFH, DM);

  // y2 = y @ W2^T  (f32 out)
  gemm_nt<0><<<dim3(DM / 128, NROWS / 128), 256, 0, stream>>>(
      yb, w2b, y2f, nullptr, NROWS, DM, FFH);

  // out = rmsnorm(y2 + alpha*x2, g2)
  resid_rms<false><<<NROWS, 256, 0, stream>>>(y2f, x2f, g2, (float*)d_out, nullptr);
}
